// PhiAttention_77713138254398
// MI455X (gfx1250) — matrix operand order repaired and checked by reading, not compiled, not run on hardware
//
#include <hip/hip_runtime.h>
#include <hip/hip_bf16.h>

typedef __attribute__((ext_vector_type(16))) __bf16 v16bf;
typedef __attribute__((ext_vector_type(8)))  __bf16 v8bf;
typedef __attribute__((ext_vector_type(8)))  float  v8f;

#define HID    2560
#define NHEADS 32
#define HDIM   80
#define HPAD   96
#define RHALF  16
#define RDIM   32
#define BATCH  2
#define SEQ    2048
#define SEQV   (SEQ + 32)   /* padded row stride for Vt so vector loads never overrun */
#define NEGBIG (-1e30f)

// D = A(16x32 bf16) * B(32x16 bf16) + C(16x16 f32)
__device__ inline v8f wmma_bf16(v16bf a, v16bf b, v8f c) {
  return __builtin_amdgcn_wmma_f32_16x16x32_bf16(false, a, false, b, (short)0, c,
                                                 false, false);
}

// Load a 16x32 bf16 A-tile operand for this lane.
// rowp points at (row base + k-chunk base) for row = lane & 15.
// element e: K = 16*(lane>=16) + e
__device__ inline v16bf load_a_tile(const __bf16* rowp, int lane) {
  int klo8 = (lane >= 16) ? 16 : 0;
  union { v16bf v; v8bf h[2]; } u;
  u.h[0] = *(const v8bf*)(rowp + klo8);
  u.h[1] = *(const v8bf*)(rowp + 8 + klo8);
  return u.v;
}

// ---------------- f32 -> bf16 elementwise ----------------
__global__ void cvt_bf16_kernel(const float* __restrict__ in,
                                __bf16* __restrict__ out, int n) {
  int i = blockIdx.x * blockDim.x + threadIdx.x;
  int stride = gridDim.x * blockDim.x;
  for (; i < n; i += stride) out[i] = (__bf16)in[i];
}

// ---------------- tiled transpose + f32->bf16: out[c][r] = in[r][c] ----------------
// R, C multiples of 32.  block (32,8), grid (C/32, R/32).
__global__ __launch_bounds__(256) void transpose_cvt_kernel(
    const float* __restrict__ in, __bf16* __restrict__ out,
    int ldin, int ldout) {
  __shared__ float tile[32][33];
  const int tx = threadIdx.x, ty = threadIdx.y;
  const int c0 = blockIdx.x * 32;
  const int r0 = blockIdx.y * 32;
  #pragma unroll
  for (int j = 0; j < 32; j += 8)
    tile[ty + j][tx] = in[(size_t)(r0 + ty + j) * ldin + (c0 + tx)];
  __syncthreads();
  #pragma unroll
  for (int j = 0; j < 32; j += 8)
    out[(size_t)(c0 + ty + j) * ldout + (r0 + tx)] = (__bf16)tile[tx][ty + j];
}

// ---------------- GEMM: C = A(bf16 MxK) * Wt(bf16 NxK, pre-transposed) + bias ----------------
// one wave computes a 32x64 output tile: 2 A-tiles x 4 B-tiles, B reused across M.
__global__ __launch_bounds__(32) void gemm_bf16_kernel(
    const __bf16* __restrict__ A, const __bf16* __restrict__ Wt,
    const float* __restrict__ bias, float* __restrict__ C,
    int M, int N, int K) {
  const int lane = threadIdx.x;
  const int half = (lane >= 16) ? 1 : 0;
  const int l15  = lane & 15;
  const int n0 = blockIdx.x * 64;
  const int m0 = blockIdx.y * 32;
  const int klo16 = half * 16;

  v8f acc[2][4] = {};
  const __bf16* arow0 = A + (size_t)(m0 + l15) * K;
  const __bf16* arow1 = arow0 + (size_t)16 * K;

  for (int k0 = 0; k0 < K; k0 += 32) {
    v16bf a0 = load_a_tile(arow0 + k0, lane);
    v16bf a1 = load_a_tile(arow1 + k0, lane);
    #pragma unroll
    for (int nt = 0; nt < 4; ++nt) {
      // B operand: lane = column (n0+nt*16+l15), elements = contiguous K
      v16bf b = *(const v16bf*)(Wt + (size_t)(n0 + nt * 16 + l15) * K + k0 + klo16);
      acc[0][nt] = wmma_bf16(a0, b, acc[0][nt]);
      acc[1][nt] = wmma_bf16(a1, b, acc[1][nt]);
    }
  }

  #pragma unroll
  for (int mi = 0; mi < 2; ++mi) {
    #pragma unroll
    for (int nt = 0; nt < 4; ++nt) {
      int col = n0 + nt * 16 + l15;
      float bi = bias[col];
      #pragma unroll
      for (int r = 0; r < 8; ++r) {
        int m = m0 + mi * 16 + r + 8 * half;
        C[(size_t)m * N + col] = acc[mi][nt][r] + bi;
      }
    }
  }
}

// ---------------- RoPE + repack q,k ----------------
// qkv: [B*S, 3*HID] f32.  Qb,Kb: [B,H,S,HPAD] bf16 (pad 80->96 with zeros).
__global__ void rope_pack_kernel(const float* __restrict__ qkv,
                                 const int* __restrict__ pos_ids,
                                 __bf16* __restrict__ Qb,
                                 __bf16* __restrict__ Kb) {
  const int t = blockIdx.x;                 // token index over B*S
  const int b = t / SEQ, s = t % SEQ;
  const float pos = (float)pos_ids[t];
  const float* row = qkv + (size_t)t * (3 * HID);
  const float LN10000 = 9.210340371976184f; // ln(10000)

  for (int idx = threadIdx.x; idx < NHEADS * HPAD; idx += blockDim.x) {
    int h = idx / HPAD, d = idx % HPAD;
    size_t obase = ((size_t)(b * NHEADS + h) * SEQ + s) * HPAD + d;
    float qv, kv;
    if (d < RDIM) {
      int i = (d < RHALF) ? d : d - RHALF;
      float ang = pos * __expf(-LN10000 * ((float)i / (float)RHALF));
      float sn, cs;
      __sincosf(ang, &sn, &cs);
      float q1 = row[h * HDIM + i];
      float q2 = row[h * HDIM + i + RHALF];
      float k1 = row[HID + h * HDIM + i];
      float k2 = row[HID + h * HDIM + i + RHALF];
      if (d < RHALF) { qv = q1 * cs - q2 * sn; kv = k1 * cs - k2 * sn; }
      else           { qv = q2 * cs + q1 * sn; kv = k2 * cs + k1 * sn; }
    } else if (d < HDIM) {
      qv = row[h * HDIM + d];
      kv = row[HID + h * HDIM + d];
    } else {
      qv = 0.f; kv = 0.f;                   // head-dim padding 80..95
    }
    Qb[obase] = (__bf16)qv;
    Kb[obase] = (__bf16)kv;
  }
}

// ---------------- causal flash attention, one wave per (b,h,16-row q tile) ----------------
// Vt: [B, HID(=h*80+d), SEQV] bf16 (d-major, seq contiguous)
__global__ __launch_bounds__(32) void flash_attn_kernel(
    const __bf16* __restrict__ Qb, const __bf16* __restrict__ Kb,
    const __bf16* __restrict__ Vt, __bf16* __restrict__ Ob) {
  __shared__ __bf16 plds[16 * 32];          // P tile staging (C layout -> A layout)
  const int lane = threadIdx.x;
  const int half = (lane >= 16) ? 1 : 0;
  const int l15  = lane & 15;
  const int q0 = blockIdx.x * 16;
  const int h  = blockIdx.y;
  const int b  = blockIdx.z;
  const float scale = 0.11180339887498949f; // 80^-0.5

  const size_t bh = (size_t)(b * NHEADS + h);
  const __bf16* Qp = Qb + bh * SEQ * HPAD;
  const __bf16* Kp = Kb + bh * SEQ * HPAD;
  const __bf16* Vp = Vt + ((size_t)b * HID + h * HDIM) * SEQV;

  // preload Q rows (16 x 96) as three A tiles
  const __bf16* qrow = Qp + (size_t)(q0 + l15) * HPAD;
  v16bf qa[3];
  #pragma unroll
  for (int c = 0; c < 3; ++c) qa[c] = load_a_tile(qrow + c * 32, lane);

  v8f acc[5] = {};
  float m[8], l[8];
  #pragma unroll
  for (int r = 0; r < 8; ++r) { m[r] = NEGBIG; l[r] = 0.f; }

  const int q_last = q0 + 15;
  const int klo16 = half * 16;
  for (int kv0 = 0; kv0 <= q_last; kv0 += 32) {
    v8f s0 = {}, s1 = {};
    // score tile 0: kv columns kv0..kv0+15
    {
      int kvcol = kv0 + l15; if (kvcol > SEQ - 1) kvcol = SEQ - 1;
      const __bf16* krow = Kp + (size_t)kvcol * HPAD + klo16;
      #pragma unroll
      for (int c = 0; c < 3; ++c)
        s0 = wmma_bf16(qa[c], *(const v16bf*)(krow + c * 32), s0);
    }
    const bool t1 = (kv0 + 16) <= q_last;
    if (t1) {
      int kvcol = kv0 + 16 + l15; if (kvcol > SEQ - 1) kvcol = SEQ - 1;
      const __bf16* krow = Kp + (size_t)kvcol * HPAD + klo16;
      #pragma unroll
      for (int c = 0; c < 3; ++c)
        s1 = wmma_bf16(qa[c], *(const v16bf*)(krow + c * 32), s1);
    }

    // scale + causal mask (C layout: row M = r + 8*half, col = lane&15)
    #pragma unroll
    for (int r = 0; r < 8; ++r) {
      int qr = q0 + r + 8 * half;
      s0[r] = ((kv0 + l15) <= qr) ? s0[r] * scale : NEGBIG;
      s1[r] = (t1 && (kv0 + 16 + l15) <= qr) ? s1[r] * scale : NEGBIG;
    }

    // online softmax; row reductions stay within each 16-lane half
    float p0[8], p1[8];
    #pragma unroll
    for (int r = 0; r < 8; ++r) {
      float v = fmaxf(s0[r], s1[r]);
      #pragma unroll
      for (int off = 1; off < 16; off <<= 1)
        v = fmaxf(v, __shfl_xor(v, off, 32));
      float mnew  = fmaxf(m[r], v);
      float alpha = __expf(m[r] - mnew);
      float e0 = __expf(s0[r] - mnew);
      float e1 = __expf(s1[r] - mnew);
      float rs = e0 + e1;
      #pragma unroll
      for (int off = 1; off < 16; off <<= 1)
        rs += __shfl_xor(rs, off, 32);
      l[r] = l[r] * alpha + rs;
      m[r] = mnew;
      p0[r] = e0; p1[r] = e1;
      #pragma unroll
      for (int d = 0; d < 5; ++d) acc[d][r] *= alpha;
    }

    // transpose P (C layout) -> A layout via LDS
    #pragma unroll
    for (int r = 0; r < 8; ++r) {
      int M = r + 8 * half;
      plds[M * 32 + l15]      = (__bf16)p0[r];
      plds[M * 32 + 16 + l15] = (__bf16)p1[r];
    }
    __syncthreads();
    union { v16bf v; v8bf hh[2]; } pu;
    {
      int klo8 = half * 16;
      pu.hh[0] = *(const v8bf*)(&plds[l15 * 32 + klo8]);
      pu.hh[1] = *(const v8bf*)(&plds[l15 * 32 + 8 + klo8]);
    }

    // PV: 5 output d-tiles; B operand = contiguous seq run of Vt row
    #pragma unroll
    for (int d = 0; d < 5; ++d) {
      int col = d * 16 + l15;   // head-dim column 0..79
      v16bf bv = *(const v16bf*)(Vp + (size_t)col * SEQV + kv0 + klo16);
      acc[d] = wmma_bf16(pu.v, bv, acc[d]);
    }
    __syncthreads();
  }

  // epilogue: normalize and write [b, q, h*80+d] bf16
  #pragma unroll
  for (int d = 0; d < 5; ++d) {
    int col = h * HDIM + d * 16 + l15;
    #pragma unroll
    for (int r = 0; r < 8; ++r) {
      int q = q0 + r + 8 * half;
      float o = acc[d][r] / l[r];
      Ob[(size_t)(b * SEQ + q) * HID + col] = (__bf16)o;
    }
  }
}

extern "C" void kernel_launch(void* const* d_in, const int* in_sizes, int n_in,
                              void* d_out, int out_size, void* d_ws, size_t ws_size,
                              hipStream_t stream) {
  const int*   pos  = (const int*)d_in[0];
  const float* hs   = (const float*)d_in[1];
  const float* Wqkv = (const float*)d_in[2];
  const float* bqkv = (const float*)d_in[3];
  const float* Wd   = (const float*)d_in[4];
  const float* bd   = (const float*)d_in[5];
  float* out = (float*)d_out;

  const size_t MTOK = (size_t)BATCH * SEQ;          // 4096
  char* p = (char*)d_ws;
  auto alloc = [&](size_t bytes) {
    char* r = p;
    p += (bytes + 255) & ~(size_t)255;
    return r;
  };
  __bf16* hsb    = (__bf16*)alloc(MTOK * HID * 2);
  __bf16* wqkvt  = (__bf16*)alloc((size_t)3 * HID * HID * 2);   // [3H][H]
  __bf16* wdt    = (__bf16*)alloc((size_t)HID * HID * 2);       // [H][H]
  float*  qkv    = (float*)alloc(MTOK * 3 * HID * 4);
  __bf16* Qbuf   = (__bf16*)alloc((size_t)BATCH * NHEADS * SEQ * HPAD * 2);
  __bf16* Kbuf   = (__bf16*)alloc((size_t)BATCH * NHEADS * SEQ * HPAD * 2);
  __bf16* Vt     = (__bf16*)alloc((size_t)BATCH * HID * SEQV * 2);
  __bf16* attnb  = (__bf16*)alloc(MTOK * HID * 2);

  cvt_bf16_kernel<<<1024, 256, 0, stream>>>(hs, hsb, (int)(MTOK * HID));

  // Wt[n][k] = W[k][n]
  transpose_cvt_kernel<<<dim3(3 * HID / 32, HID / 32), dim3(32, 8), 0, stream>>>(
      Wqkv, wqkvt, 3 * HID, HID);
  transpose_cvt_kernel<<<dim3(HID / 32, HID / 32), dim3(32, 8), 0, stream>>>(
      Wd, wdt, HID, HID);

  gemm_bf16_kernel<<<dim3(3 * HID / 64, MTOK / 32), 32, 0, stream>>>(
      hsb, wqkvt, bqkv, qkv, (int)MTOK, 3 * HID, HID);

  rope_pack_kernel<<<(int)MTOK, 256, 0, stream>>>(qkv, pos, Qbuf, Kbuf);

  // Vt[b][hd][s] = qkv[b*SEQ+s][2*HID + hd]  (per-batch tiled transpose)
  for (int b = 0; b < BATCH; ++b) {
    transpose_cvt_kernel<<<dim3(HID / 32, SEQ / 32), dim3(32, 8), 0, stream>>>(
        qkv + (size_t)b * SEQ * 3 * HID + 2 * HID,
        Vt + (size_t)b * HID * SEQV, 3 * HID, SEQV);
  }

  flash_attn_kernel<<<dim3(SEQ / 16, NHEADS, BATCH), 32, 0, stream>>>(
      Qbuf, Kbuf, Vt, attnb);

  gemm_bf16_kernel<<<dim3(HID / 64, MTOK / 32), 32, 0, stream>>>(
      attnb, wdt, bd, out, (int)MTOK, HID, HID);
}
